// VisualAdapter_9320079032662
// MI455X (gfx1250) — matrix operand order repaired and checked by reading, not compiled, not run on hardware
//
#include <hip/hip_runtime.h>
#include <hip/hip_bf16.h>

// ---------------------------------------------------------------------------
// VisualAdapter for MI455X (gfx1250, wave32, WMMA, LDS-staged GEMM).
// GEMM: 64x64 block tile, K-step 32, f32->f16 staged through LDS,
// ds_load_b128 fragment reads, v_wmma_f32_16x16x32_f16 accumulate.
// Edge handling: clamped unconditional loads laundered through one empty
// asm volatile (prevents predication-sinking without volatile's per-load
// SYS-scope serialization), then one v_cndmask per element. The fast/slow
// choice is block-uniform so it lowers to a scalar branch.
// ---------------------------------------------------------------------------

typedef __attribute__((ext_vector_type(16))) _Float16 v16h;
typedef __attribute__((ext_vector_type(8)))  float    v8f;
typedef __attribute__((ext_vector_type(4)))  float    f4;
typedef __attribute__((ext_vector_type(8)))  _Float16 h8v;

union Frag16 { v16h v; h8v v8[2]; _Float16 h[16]; };
union CFrag  { v8f  v; float f[8]; };
union H8     { h8v  v; _Float16 h[8]; };

#define ACT_NONE 0
#define ACT_RELU 1
#define ACT_SIG  2

#define LDS_STRIDE 40   // halfs per row (80B, multiple of 16B, conflict-spreading)

// Force the 8 loaded values to materialize unconditionally (single wait),
// so the compiler cannot sink the loads under an EXEC predicate.
#define LAUNDER8(a) asm volatile("" : "+v"((a)[0]), "+v"((a)[1]), "+v"((a)[2]), "+v"((a)[3]), \
                                       "+v"((a)[4]), "+v"((a)[5]), "+v"((a)[6]), "+v"((a)[7]))

// -------------------------------- WMMA GEMM --------------------------------
// C[b] = act( alpha * A[b] x opB(B[b]) (+ C_old) (+ bias) (+ Cadd) )
// A: [M,K] row-major (optionally scaled per-K by (1+ascale[k])).
// B: transB=1 -> stored [N,K] row-major; transB=0 -> stored [K,N] row-major.
__global__ __launch_bounds__(256) void wmma_gemm_f32(
    const float* __restrict__ A, const float* __restrict__ Bm, float* __restrict__ C,
    const float* __restrict__ bias, const float* __restrict__ Cadd,
    const float* __restrict__ ascale, const float* __restrict__ alpha_ptr,
    int M, int N, int K, int lda, int ldb, int ldc,
    long long sA, long long sB, long long sC, long long sAscale,
    int ldcadd, long long sCadd,
    int transB, int act, int accumulate, int biasPerRow)
{
    __shared__ _Float16 lsA[64 * LDS_STRIDE];  // A tile: [row 0..63][k 0..31]
    __shared__ _Float16 lsB[64 * LDS_STRIDE];  // B tile: [col 0..63][k 0..31]

    const int b = blockIdx.z;
    const float* __restrict__ Ab = A  + (long long)b * sA;
    const float* __restrict__ Bb = Bm + (long long)b * sB;
    float*       __restrict__ Cb = C  + (long long)b * sC;
    const float* __restrict__ scl = ascale ? (ascale + (long long)b * sAscale) : nullptr;

    const int t    = threadIdx.x;
    const int lane = t & 31;
    const int wave = t >> 5;
    const int wm = wave >> 1;                 // 0..3 : 16-row slab within 64
    const int wn = wave & 1;                  // 0..1 : 32-col slab within 64
    const int blockM = blockIdx.y * 64;
    const int blockN = blockIdx.x * 64;

    const int half = lane >> 4;
    const int lr   = lane & 15;

    // cooperative-load coordinates
    const int arow = t >> 2;                  // 0..63
    const int akc  = (t & 3) << 3;            // 0,8,16,24
    const int krow = t >> 3;                  // non-trans path: k, 0..31
    const int bnc  = (t & 7) << 3;            // non-trans path: n chunk, 0..56

    // block-uniform full-tile flags (lower to scalar branches, no EXEC churn)
    const bool fullM  = (blockM + 64 <= M);
    const bool fullNB = (blockN + 64 <= N);

    CFrag c0, c1;
    #pragma unroll
    for (int i = 0; i < 8; ++i) { c0.f[i] = 0.0f; c1.f[i] = 0.0f; }

    for (int kk = 0; kk < K; kk += 32) {
        const bool fullK = (kk + 32 <= K);

        // ---- stage A tile (64 x 32) into LDS, f32 -> f16, fused per-K scale
        {
            const int gm0 = blockM + arow;
            const long long rowoff = (long long)(gm0 < M ? gm0 : (M - 1)) * lda;
            float tmp[8];
            if (fullM && fullK) {                      // uniform fast path
                const float* src = Ab + rowoff + kk + akc;
                f4 f0 = *(const f4*)src;
                f4 f1 = *(const f4*)(src + 4);
                tmp[0]=f0[0]; tmp[1]=f0[1]; tmp[2]=f0[2]; tmp[3]=f0[3];
                tmp[4]=f1[0]; tmp[5]=f1[1]; tmp[6]=f1[2]; tmp[7]=f1[3];
                if (kk + 32 < K) __builtin_prefetch(src + 32, 0, 1);
            } else {                                   // branchless edge path
                #pragma unroll
                for (int j = 0; j < 8; ++j) {
                    const int k2  = kk + akc + j;
                    const int kc2 = k2 < K ? k2 : (K - 1);
                    tmp[j] = Ab[rowoff + kc2];         // clamped, always legal
                }
                LAUNDER8(tmp);
                #pragma unroll
                for (int j = 0; j < 8; ++j) {
                    const int k2 = kk + akc + j;
                    tmp[j] = (gm0 < M && k2 < K) ? tmp[j] : 0.0f;
                }
            }
            if (scl) {
                #pragma unroll
                for (int j = 0; j < 8; ++j) {
                    const int k2  = kk + akc + j;
                    const int kc2 = k2 < K ? k2 : (K - 1);
                    tmp[j] *= (1.0f + scl[kc2]);       // tmp==0 when OOB
                }
            }
            H8 hv;
            #pragma unroll
            for (int j = 0; j < 8; ++j) hv.h[j] = (_Float16)tmp[j];
            *(h8v*)&lsA[arow * LDS_STRIDE + akc] = hv.v;   // ds_store_b128
        }

        // ---- stage B tile into LDS as [col][k]
        if (transB) {
            const int gn0 = blockN + arow;
            const long long rowoff = (long long)(gn0 < N ? gn0 : (N - 1)) * ldb;
            float tmp[8];
            if (fullNB && fullK) {
                const float* src = Bb + rowoff + kk + akc;
                f4 f0 = *(const f4*)src;
                f4 f1 = *(const f4*)(src + 4);
                tmp[0]=f0[0]; tmp[1]=f0[1]; tmp[2]=f0[2]; tmp[3]=f0[3];
                tmp[4]=f1[0]; tmp[5]=f1[1]; tmp[6]=f1[2]; tmp[7]=f1[3];
                if (kk + 32 < K) __builtin_prefetch(src + 32, 0, 1);
            } else {
                #pragma unroll
                for (int j = 0; j < 8; ++j) {
                    const int k2  = kk + akc + j;
                    const int kc2 = k2 < K ? k2 : (K - 1);
                    tmp[j] = Bb[rowoff + kc2];
                }
                LAUNDER8(tmp);
                #pragma unroll
                for (int j = 0; j < 8; ++j) {
                    const int k2 = kk + akc + j;
                    tmp[j] = (gn0 < N && k2 < K) ? tmp[j] : 0.0f;
                }
            }
            H8 hv;
            #pragma unroll
            for (int j = 0; j < 8; ++j) hv.h[j] = (_Float16)tmp[j];
            *(h8v*)&lsB[arow * LDS_STRIDE + akc] = hv.v;
        } else {
            const int gk0 = kk + krow;
            const long long rowoff = (long long)(gk0 < K ? gk0 : (K - 1)) * ldb;
            float tmp[8];
            if (fullNB && fullK) {
                const float* src = Bb + rowoff + blockN + bnc;
                f4 f0 = *(const f4*)src;
                f4 f1 = *(const f4*)(src + 4);
                tmp[0]=f0[0]; tmp[1]=f0[1]; tmp[2]=f0[2]; tmp[3]=f0[3];
                tmp[4]=f1[0]; tmp[5]=f1[1]; tmp[6]=f1[2]; tmp[7]=f1[3];
            } else {
                #pragma unroll
                for (int j = 0; j < 8; ++j) {
                    const int n2  = blockN + bnc + j;
                    const int nc2 = n2 < N ? n2 : (N - 1);
                    tmp[j] = Bb[rowoff + nc2];
                }
                LAUNDER8(tmp);
                #pragma unroll
                for (int j = 0; j < 8; ++j) {
                    const int n2 = blockN + bnc + j;
                    tmp[j] = (gk0 < K && n2 < N) ? tmp[j] : 0.0f;
                }
            }
            #pragma unroll
            for (int j = 0; j < 8; ++j)        // transpose into [col][k]
                lsB[(bnc + j) * LDS_STRIDE + krow] = (_Float16)tmp[j];
        }
        __syncthreads();

        // ---- fragments from LDS (all b128, 16B aligned, EXEC all-ones)
        // A 16x32: lane=row, K = half*16 + e (contiguous 16)
        Frag16 af;
        const int ar = (wm << 4) + lr;
        af.v8[0] = *(const h8v*)&lsA[ar * LDS_STRIDE + (half << 4)];
        af.v8[1] = *(const h8v*)&lsA[ar * LDS_STRIDE + (half << 4) + 8];
        // B 32x16: lane=col, K = half*16 + e (contiguous 16)
        Frag16 bf0, bf1;
        const int bc0 = (wn << 5) + lr;
        const int bc1 = bc0 + 16;
        bf0.v8[0] = *(const h8v*)&lsB[bc0 * LDS_STRIDE + (half << 4)];
        bf0.v8[1] = *(const h8v*)&lsB[bc0 * LDS_STRIDE + (half << 4) + 8];
        bf1.v8[0] = *(const h8v*)&lsB[bc1 * LDS_STRIDE + (half << 4)];
        bf1.v8[1] = *(const h8v*)&lsB[bc1 * LDS_STRIDE + (half << 4) + 8];

        c0.v = __builtin_amdgcn_wmma_f32_16x16x32_f16(false, af.v, false, bf0.v,
                                                      (short)0, c0.v, false, false);
        c1.v = __builtin_amdgcn_wmma_f32_16x16x32_f16(false, af.v, false, bf1.v,
                                                      (short)0, c1.v, false, false);
        __syncthreads();
    }

    const float alpha = alpha_ptr ? alpha_ptr[0] : 1.0f;
    // D layout: lane=col (n = lane%16), VGPR r -> row m = r + 8*(lane/16)
    #pragma unroll
    for (int s2 = 0; s2 < 2; ++s2) {
        const CFrag& cc = s2 ? c1 : c0;
        const int n = blockN + (wn << 5) + (s2 << 4) + lr;
        if (n < N) {
            #pragma unroll
            for (int r = 0; r < 8; ++r) {
                const int m = blockM + (wm << 4) + r + (half << 3);
                if (m < M) {
                    float v = cc.f[r] * alpha;
                    if (accumulate) v += Cb[(long long)m * ldc + n];
                    if (bias)       v += biasPerRow ? bias[m] : bias[n];
                    if (Cadd)       v += (Cadd + (long long)b * sCadd)[(long long)m * ldcadd + n];
                    if (act == ACT_RELU)     v = v > 0.0f ? v : 0.0f;
                    else if (act == ACT_SIG) v = 1.0f / (1.0f + __expf(-v));
                    Cb[(long long)m * ldc + n] = v;
                }
            }
        }
    }
}

// ---------------------------- auxiliary kernels ----------------------------

__global__ __launch_bounds__(256) void k_transpose(
    const float* __restrict__ in, float* __restrict__ out, int Dd, int Nn, long long total)
{   // in [B,D,N] -> out [B,N,D]
    long long idx = (long long)blockIdx.x * 256 + threadIdx.x;
    if (idx >= total) return;
    int n = (int)(idx % Nn); long long t = idx / Nn;
    int d = (int)(t % Dd);   int b = (int)(t / Dd);
    out[((long long)b * Nn + n) * Dd + d] = in[idx];
}

__global__ __launch_bounds__(256) void k_softmax(float* __restrict__ data, int C)
{   // one block per row, softmax over last dim, in place
    float* p = data + (long long)blockIdx.x * C;
    __shared__ float red[256];
    int tid = threadIdx.x;
    float mx = -1e30f;
    for (int i = tid; i < C; i += 256) mx = fmaxf(mx, p[i]);
    red[tid] = mx; __syncthreads();
    for (int s = 128; s > 0; s >>= 1) { if (tid < s) red[tid] = fmaxf(red[tid], red[tid + s]); __syncthreads(); }
    mx = red[0]; __syncthreads();
    float sum = 0.0f;
    for (int i = tid; i < C; i += 256) { float e = __expf(p[i] - mx); p[i] = e; sum += e; }
    red[tid] = sum; __syncthreads();
    for (int s = 128; s > 0; s >>= 1) { if (tid < s) red[tid] += red[tid + s]; __syncthreads(); }
    float inv = 1.0f / red[0];
    for (int i = tid; i < C; i += 256) p[i] *= inv;
}

__global__ __launch_bounds__(256) void k_colmean(
    const float* __restrict__ in, float* __restrict__ out,
    const float* __restrict__ mul, int Mrows, int Dc)
{   // out[b,d] = mean_m in[b,m,d] (* mul[b,d])
    int d = blockIdx.x * 256 + threadIdx.x;
    int b = blockIdx.y;
    if (d >= Dc) return;
    const float* p = in + (long long)b * Mrows * Dc + d;
    float s = 0.0f;
    for (int m = 0; m < Mrows; ++m) s += p[(long long)m * Dc];
    s *= (1.0f / (float)Mrows);
    if (mul) s *= mul[(long long)b * Dc + d];
    out[(long long)b * Dc + d] = s;
}

__global__ __launch_bounds__(256) void k_stmp(
    const float* __restrict__ cq, const float* __restrict__ aq2,
    const float* __restrict__ vsw, const float* __restrict__ vsb,
    float* __restrict__ stmp, float* __restrict__ ssig, int Nn, int DMc, int total)
{   // s_tmp[b,n] = sum_e cq[b,n,e]*aq2[b,e]*vs_w[e] + vs_b ; s_sig = sigmoid
    int idx = blockIdx.x * 256 + threadIdx.x;
    if (idx >= total) return;
    int b = idx / Nn;
    const float* row = cq  + (long long)idx * DMc;
    const float* a   = aq2 + (long long)b   * DMc;
    float s = vsb[0];
    for (int e = 0; e < DMc; ++e) s += row[e] * a[e] * vsw[e];
    stmp[idx] = s;
    ssig[idx] = 1.0f / (1.0f + __expf(-s));
}

__global__ __launch_bounds__(256) void k_spatt(
    const float* __restrict__ stmp, float* __restrict__ out, int Nn)
{   // sp_att[b,:] = softmax_n tanh(s_tmp[b,:])   (Nn <= 256)
    int b = blockIdx.x, tid = threadIdx.x;
    const float* p = stmp + (long long)b * Nn;
    __shared__ float red[256];
    float t = (tid < Nn) ? tanhf(p[tid]) : -1e30f;
    red[tid] = t; __syncthreads();
    for (int s = 128; s > 0; s >>= 1) { if (tid < s) red[tid] = fmaxf(red[tid], red[tid + s]); __syncthreads(); }
    float mx = red[0]; __syncthreads();
    float e = (tid < Nn) ? __expf(t - mx) : 0.0f;
    red[tid] = e; __syncthreads();
    for (int s = 128; s > 0; s >>= 1) { if (tid < s) red[tid] += red[tid + s]; __syncthreads(); }
    if (tid < Nn) out[(long long)b * Nn + tid] = e / red[0];
}

__global__ __launch_bounds__(256) void k_gate_xT(
    float* __restrict__ xT, const float* __restrict__ ch,
    const float* __restrict__ ssig, int Nn, int Dd, long long total)
{   // xT *= (0.3*ch_att[b,d] + 0.05*s_sig[b,n] + 0.7)
    long long idx = (long long)blockIdx.x * 256 + threadIdx.x;
    if (idx >= total) return;
    int d = (int)(idx % Dd); long long t = idx / Dd;
    int n = (int)(t % Nn);   int b = (int)(t / Nn);
    xT[idx] *= (0.3f * ch[(long long)b * Dd + d] + 0.05f * ssig[(long long)b * Nn + n] + 0.7f);
}

__global__ __launch_bounds__(256) void k_layernorm(
    float* __restrict__ data, const float* __restrict__ g, const float* __restrict__ bta, int C)
{   // one block per row, in place
    float* p = data + (long long)blockIdx.x * C;
    __shared__ float red[256];
    int tid = threadIdx.x;
    float s = 0.0f;
    for (int i = tid; i < C; i += 256) s += p[i];
    red[tid] = s; __syncthreads();
    for (int st = 128; st > 0; st >>= 1) { if (tid < st) red[tid] += red[tid + st]; __syncthreads(); }
    float mean = red[0] / (float)C; __syncthreads();
    float v = 0.0f;
    for (int i = tid; i < C; i += 256) { float d = p[i] - mean; v += d * d; }
    red[tid] = v; __syncthreads();
    for (int st = 128; st > 0; st >>= 1) { if (tid < st) red[tid] += red[tid + st]; __syncthreads(); }
    float rs = rsqrtf(red[0] / (float)C + 1e-5f);
    for (int i = tid; i < C; i += 256) p[i] = (p[i] - mean) * rs * g[i] + bta[i];
}

__global__ __launch_bounds__(256) void k_bnstats(
    const float* __restrict__ data, float* __restrict__ stats, int rows, int C)
{   // per-channel mean/var over rows; data [rows, C]; stats [2*C]
    int c = blockIdx.x, tid = threadIdx.x;
    float s = 0.0f, s2 = 0.0f;
    for (int r = tid; r < rows; r += 256) { float v = data[(long long)r * C + c]; s += v; s2 += v * v; }
    __shared__ float rs_[256]; __shared__ float rq_[256];
    rs_[tid] = s; rq_[tid] = s2; __syncthreads();
    for (int st = 128; st > 0; st >>= 1) {
        if (tid < st) { rs_[tid] += rs_[tid + st]; rq_[tid] += rq_[tid + st]; }
        __syncthreads();
    }
    if (tid == 0) { float mu = rs_[0] / rows; stats[c] = mu; stats[C + c] = rq_[0] / rows - mu * mu; }
}

__global__ __launch_bounds__(256) void k_bn_relu(
    float* __restrict__ data, const float* __restrict__ stats,
    const float* __restrict__ g, const float* __restrict__ bta, long long total, int C)
{
    long long idx = (long long)blockIdx.x * 256 + threadIdx.x;
    if (idx >= total) return;
    int c = (int)(idx % C);
    float v = (data[idx] - stats[c]) * rsqrtf(stats[C + c] + 1e-5f) * g[c] + bta[c];
    data[idx] = fmaxf(v, 0.0f);
}

__global__ __launch_bounds__(256) void k_final(
    const float* __restrict__ up, const float* __restrict__ st2,
    const float* __restrict__ g2, const float* __restrict__ b2,
    const float* __restrict__ lg, const float* __restrict__ lb,
    const float* __restrict__ gate, float* __restrict__ out, int Nn, int Dd)
{   // per (b,n): h = BN2(up_row); y = LN(h)*lnp; out[b,d,n] = gate*y   (Dd == 768 == 3*256)
    int r = blockIdx.x, tid = threadIdx.x;
    int b = r / Nn, n = r % Nn;
    const float* row = up + (long long)r * Dd;
    __shared__ float red[256];
    float h[3]; float s = 0.0f;
    #pragma unroll
    for (int j = 0; j < 3; ++j) {
        int d = tid + j * 256;
        float v = (row[d] - st2[d]) * rsqrtf(st2[Dd + d] + 1e-5f) * g2[d] + b2[d];
        h[j] = v; s += v;
    }
    red[tid] = s; __syncthreads();
    for (int st = 128; st > 0; st >>= 1) { if (tid < st) red[tid] += red[tid + st]; __syncthreads(); }
    float mean = red[0] / (float)Dd; __syncthreads();
    float v2 = 0.0f;
    #pragma unroll
    for (int j = 0; j < 3; ++j) { float d_ = h[j] - mean; v2 += d_ * d_; }
    red[tid] = v2; __syncthreads();
    for (int st = 128; st > 0; st >>= 1) { if (tid < st) red[tid] += red[tid + st]; __syncthreads(); }
    float rs = rsqrtf(red[0] / (float)Dd + 1e-5f);
    float gv = gate[0];
    #pragma unroll
    for (int j = 0; j < 3; ++j) {
        int d = tid + j * 256;
        out[((long long)b * Dd + d) * (long long)Nn + n] = ((h[j] - mean) * rs * lg[d] + lb[d]) * gv;
    }
}

// ------------------------------- host driver -------------------------------

static inline void launch_gemm(hipStream_t s,
    const float* A, const float* B, float* C,
    const float* bias, const float* Cadd, const float* ascale, const float* alpha_ptr,
    int M, int N, int K, int lda, int ldb, int ldc,
    long long sA, long long sB, long long sC, long long sAscale,
    int ldcadd, long long sCadd, int batch,
    int transB, int act, int accumulate, int biasPerRow)
{
    dim3 grid((N + 63) / 64, (M + 63) / 64, batch);
    wmma_gemm_f32<<<grid, 256, 0, s>>>(A, B, C, bias, Cadd, ascale, alpha_ptr,
        M, N, K, lda, ldb, ldc, sA, sB, sC, sAscale, ldcadd, sCadd,
        transB, act, accumulate, biasPerRow);
}

extern "C" void kernel_launch(void* const* d_in, const int* in_sizes, int n_in,
                              void* d_out, int out_size, void* d_ws, size_t ws_size,
                              hipStream_t stream) {
    (void)in_sizes; (void)n_in; (void)out_size; (void)ws_size;

    const int Bn = 64, Dd = 768, Nn = 196, LINc = 1536, CINc = 196, COUTc = 196;
    const int Tt = 32, DMc = 384, DDc = 96;

    const float* x       = (const float*)d_in[0];
    const float* vis     = (const float*)d_in[1];
    const float* conv_w  = (const float*)d_in[2];
    const float* conv_b  = (const float*)d_in[3];
    const float* fc_w    = (const float*)d_in[4];
    const float* fc_b    = (const float*)d_in[5];
    const float* a1_w    = (const float*)d_in[6];
    const float* a1_b    = (const float*)d_in[7];
    const float* v1_w    = (const float*)d_in[8];
    const float* v1_b    = (const float*)d_in[9];
    const float* btl_w   = (const float*)d_in[10];
    const float* btl_b   = (const float*)d_in[11];
    const float* v2_w    = (const float*)d_in[12];
    const float* v2_b    = (const float*)d_in[13];
    const float* a2_w    = (const float*)d_in[14];
    const float* a2_b    = (const float*)d_in[15];
    const float* vs_w    = (const float*)d_in[16];
    const float* vs_b    = (const float*)d_in[17];
    const float* vc_w    = (const float*)d_in[18];
    const float* vc_b    = (const float*)d_in[19];
    const float* my_tok  = (const float*)d_in[20];
    const float* gate_av = (const float*)d_in[21];
    const float* gate    = (const float*)d_in[22];
    const float* down_w  = (const float*)d_in[23];
    const float* up_w    = (const float*)d_in[24];
    const float* bn1_g   = (const float*)d_in[25];
    const float* bn1_b   = (const float*)d_in[26];
    const float* bn2_g   = (const float*)d_in[27];
    const float* bn2_b   = (const float*)d_in[28];
    const float* lnb_g   = (const float*)d_in[29];
    const float* lnb_b   = (const float*)d_in[30];
    const float* lnp_g   = (const float*)d_in[31];
    const float* lnp_b   = (const float*)d_in[32];
    float* out = (float*)d_out;

    // ---- workspace layout (~170 MB of f32; big buffers re-used) ----
    float* wsf = (float*)d_ws;
    size_t off = 0;
    auto take = [&](size_t n) { float* p = wsf + off; off += (n + 63) & ~(size_t)63; return p; };
    float* buf_vtc   = take((size_t)Bn * COUTc * LINc);  // conv out; reused for vq1, c_q
    float* buf_vtf   = take((size_t)Bn * COUTc * Dd);    // fc out;   reused for up-proj out
    float* buf_xT    = take((size_t)Bn * Nn * Dd);
    float* buf_att1  = take((size_t)Bn * Tt * COUTc);
    float* buf_rep   = take((size_t)Bn * Tt * Dd);
    float* buf_att2  = take((size_t)Bn * Nn * Tt);
    float* buf_audio = take((size_t)Bn * Dd);
    float* buf_aq1   = take((size_t)Bn * Dd);
    float* buf_avqin = take((size_t)Bn * Dd);
    float* buf_avq   = take((size_t)Bn * DMc);
    float* buf_chatt = take((size_t)Bn * Dd);
    float* buf_aq2   = take((size_t)Bn * DMc);
    float* buf_stmp  = take((size_t)Bn * Nn);
    float* buf_ssig  = take((size_t)Bn * Nn);
    float* buf_z     = take((size_t)Bn * Nn * DDc);
    float* buf_bn1   = take((size_t)2 * DDc);
    float* buf_bn2   = take((size_t)2 * Dd);
    float* buf_vq1 = buf_vtc;
    float* buf_cq  = buf_vtc;
    float* buf_up  = buf_vtf;

    // 1) vtc[b] = conv_w @ vis[b]^T + conv_b(rows)   [COUT x LIN]
    launch_gemm(stream, conv_w, vis, buf_vtc, conv_b, nullptr, nullptr, nullptr,
                COUTc, LINc, CINc, CINc, CINc, LINc,
                0, (long long)LINc * CINc, (long long)COUTc * LINc, 0, 0, 0,
                Bn, /*transB=*/1, ACT_NONE, 0, /*biasPerRow=*/1);

    // 2) vtf[b] = vtc[b] @ fc_w^T + fc_b             [COUT x D]
    launch_gemm(stream, buf_vtc, fc_w, buf_vtf, fc_b, nullptr, nullptr, nullptr,
                COUTc, Dd, LINc, LINc, LINc, Dd,
                (long long)COUTc * LINc, 0, (long long)COUTc * Dd, 0, 0, 0,
                Bn, 1, ACT_NONE, 0, 0);

    // 3) att1[b] = my_tokens @ vtf[b]^T              [T x COUT]
    launch_gemm(stream, my_tok, buf_vtf, buf_att1, nullptr, nullptr, nullptr, nullptr,
                Tt, COUTc, Dd, Dd, Dd, COUTc,
                0, (long long)COUTc * Dd, (long long)Tt * COUTc, 0, 0, 0,
                Bn, 1, ACT_NONE, 0, 0);
    k_softmax<<<Bn * Tt, 256, 0, stream>>>(buf_att1, COUTc);

    // 4) rep[b] = att1[b] @ vtf[b] + my_tokens       [T x D]
    launch_gemm(stream, buf_att1, buf_vtf, buf_rep, nullptr, my_tok, nullptr, nullptr,
                Tt, Dd, COUTc, COUTc, Dd, Dd,
                (long long)Tt * COUTc, (long long)COUTc * Dd, (long long)Tt * Dd, 0,
                Dd, 0, Bn, /*transB=*/0, ACT_NONE, 0, 0);

    // 5) xT = transpose(x)                           [B,N,D]
    {
        long long total = (long long)Bn * Dd * Nn;
        k_transpose<<<(int)((total + 255) / 256), 256, 0, stream>>>(x, buf_xT, Dd, Nn, total);
    }

    // 6) att2[b] = xT[b] @ rep[b]^T                  [N x T]
    launch_gemm(stream, buf_xT, buf_rep, buf_att2, nullptr, nullptr, nullptr, nullptr,
                Nn, Tt, Dd, Dd, Dd, Tt,
                (long long)Nn * Dd, (long long)Tt * Dd, (long long)Nn * Tt, 0, 0, 0,
                Bn, 1, ACT_NONE, 0, 0);
    k_softmax<<<Bn * Nn, 256, 0, stream>>>(buf_att2, Tt);

    // 7) xT += gate_av * att2[b] @ rep[b]
    launch_gemm(stream, buf_att2, buf_rep, buf_xT, nullptr, nullptr, nullptr, gate_av,
                Nn, Dd, Tt, Tt, Dd, Dd,
                (long long)Nn * Tt, (long long)Tt * Dd, (long long)Nn * Dd, 0, 0, 0,
                Bn, /*transB=*/0, ACT_NONE, /*accumulate=*/1, 0);

    // 8) audio[b,d] = mean_m vtf[b,m,d]
    k_colmean<<<dim3((Dd + 255) / 256, Bn), 256, 0, stream>>>(buf_vtf, buf_audio, nullptr, COUTc, Dd);

    // 9) aq1 = relu(audio @ a1_w^T + a1_b)           [B x D], single GEMM
    launch_gemm(stream, buf_audio, a1_w, buf_aq1, a1_b, nullptr, nullptr, nullptr,
                Bn, Dd, Dd, Dd, Dd, Dd, 0, 0, 0, 0, 0, 0, 1, 1, ACT_RELU, 0, 0);

    // 10) vq1[b] = relu(xT[b] @ v1_w^T + v1_b)       [N x D]  (into vtc region)
    launch_gemm(stream, buf_xT, v1_w, buf_vq1, v1_b, nullptr, nullptr, nullptr,
                Nn, Dd, Dd, Dd, Dd, Dd,
                (long long)Nn * Dd, 0, (long long)Nn * Dd, 0, 0, 0,
                Bn, 1, ACT_RELU, 0, 0);

    // 11) avq_in[b,d] = aq1[b,d] * mean_n vq1[b,n,d]
    k_colmean<<<dim3((Dd + 255) / 256, Bn), 256, 0, stream>>>(buf_vq1, buf_avqin, buf_aq1, Nn, Dd);

    // 12) avq = relu(avq_in @ btl_w^T + btl_b)       [B x DM]
    launch_gemm(stream, buf_avqin, btl_w, buf_avq, btl_b, nullptr, nullptr, nullptr,
                Bn, DMc, Dd, Dd, Dd, DMc, 0, 0, 0, 0, 0, 0, 1, 1, ACT_RELU, 0, 0);

    // 13) ch_att = sigmoid(avq @ vc_w^T + vc_b)      [B x D]
    launch_gemm(stream, buf_avq, vc_w, buf_chatt, vc_b, nullptr, nullptr, nullptr,
                Bn, Dd, DMc, DMc, DMc, Dd, 0, 0, 0, 0, 0, 0, 1, 1, ACT_SIG, 0, 0);

    // 14) aq2 = relu(audio @ a2_w^T + a2_b)          [B x DM]
    launch_gemm(stream, buf_audio, a2_w, buf_aq2, a2_b, nullptr, nullptr, nullptr,
                Bn, DMc, Dd, Dd, Dd, DMc, 0, 0, 0, 0, 0, 0, 1, 1, ACT_RELU, 0, 0);

    // 15) c_q[b] = relu( (xT[b] * (1+ch_att[b,:])) @ v2_w^T + v2_b )  [N x DM]
    launch_gemm(stream, buf_xT, v2_w, buf_cq, v2_b, nullptr, buf_chatt, nullptr,
                Nn, DMc, Dd, Dd, Dd, DMc,
                (long long)Nn * Dd, 0, (long long)Nn * DMc, (long long)Dd, 0, 0,
                Bn, 1, ACT_RELU, 0, 0);

    // 16) s_tmp / s_sig, and sp_att (2nd output)
    k_stmp<<<(Bn * Nn + 255) / 256, 256, 0, stream>>>(buf_cq, buf_aq2, vs_w, vs_b,
                                                      buf_stmp, buf_ssig, Nn, DMc, Bn * Nn);
    k_spatt<<<Bn, 256, 0, stream>>>(buf_stmp, out + (long long)Bn * Dd * Nn, Nn);

    // 17) xT *= (0.3*ch_att + 0.05*s_sig + 0.7)
    {
        long long total = (long long)Bn * Nn * Dd;
        k_gate_xT<<<(int)((total + 255) / 256), 256, 0, stream>>>(buf_xT, buf_chatt, buf_ssig, Nn, Dd, total);
    }

    // 18) LayerNorm (ln_before) over D, in place
    k_layernorm<<<Bn * Nn, 256, 0, stream>>>(buf_xT, lnb_g, lnb_b, Dd);

    // 19) down: z[b] = xT[b] @ down_w^T              [N x DD]
    launch_gemm(stream, buf_xT, down_w, buf_z, nullptr, nullptr, nullptr, nullptr,
                Nn, DDc, Dd, Dd, Dd, DDc,
                (long long)Nn * Dd, 0, (long long)Nn * DDc, 0, 0, 0,
                Bn, 1, ACT_NONE, 0, 0);

    // 20) BN1 (training stats over B*N) + ReLU
    k_bnstats<<<DDc, 256, 0, stream>>>(buf_z, buf_bn1, Bn * Nn, DDc);
    {
        long long total = (long long)Bn * Nn * DDc;
        k_bn_relu<<<(int)((total + 255) / 256), 256, 0, stream>>>(buf_z, buf_bn1, bn1_g, bn1_b, total, DDc);
    }

    // 21) up: up_out[b] = z[b] @ up_w^T              [N x D]  (into vtf region)
    launch_gemm(stream, buf_z, up_w, buf_up, nullptr, nullptr, nullptr, nullptr,
                Nn, Dd, DDc, DDc, DDc, Dd,
                (long long)Nn * DDc, 0, (long long)Nn * Dd, 0, 0, 0,
                Bn, 1, ACT_NONE, 0, 0);

    // 22) BN2 stats, then fused BN2 + LN(lnp) + gate + transposed store -> out[B,D,N]
    k_bnstats<<<Dd, 256, 0, stream>>>(buf_up, buf_bn2, Bn * Nn, Dd);
    k_final<<<Bn * Nn, 256, 0, stream>>>(buf_up, buf_bn2, bn2_g, bn2_b, lnp_g, lnp_b, gate, out, Nn, Dd);
}
